// SimpleBiLSTMSentenceEncoder_8813272891995
// MI455X (gfx1250) — matrix operand order repaired and checked by reading, not compiled, not run on hardware
//
#include <hip/hip_runtime.h>
#include <hip/hip_bf16.h>
#include <math.h>

// ---------------------------------------------------------------------------
// BiLSTM sentence encoder for gfx1250 (MI455X).
// bf16 WMMA + fp32 cell state. B-tiles staged to LDS with
// GLOBAL_LOAD_ASYNC_TO_LDS_B128 (ASYNCcnt), triple-buffered 16KB stages
// (64 K-elems each, 8 WMMA per barrier), stage loop unrolled by 3.
// DS-read / WMMA interleave enforced with sched_group_barrier.
// T=128, B=64, E=300 (padded 320), H=2048, 4H=8192.
// ---------------------------------------------------------------------------

#define T_LEN 128
#define BSZ   64
#define EDIM  300
#define EPAD  320
#define HDIM  2048
#define GDIM  8192   // 4*H

#define ST_H  (HDIM / 64)          // 32 stages over h (64 K-elems each)
#define ST_X  (EPAD / 64)          // 5 stages over x
#define ST_N  (ST_H + ST_X)        // 37 total

#define LDS_STAGE 16384            // bytes/buffer: 2 khalves * 2 jtiles * 4 gates * 16 rows * 64B

typedef __attribute__((ext_vector_type(16))) __bf16 v16bf;
typedef __attribute__((ext_vector_type(8)))  float  v8f;

union Frag { v16bf v; uint4 q[2]; };

#if __has_builtin(__builtin_amdgcn_sched_group_barrier)
#define SGB(mask, size, id) __builtin_amdgcn_sched_group_barrier((mask), (size), (id))
#else
#define SGB(mask, size, id)
#endif
#define SGB_DSREAD 0x100
#define SGB_MATMUL 0x008

__device__ __forceinline__ float sigmoidf_(float x) {
    return 1.0f / (1.0f + __expf(-x));
}

__device__ __forceinline__ void wait_async(bool pending) {
    // async loads complete in order (ISA 08 §4.1); each stage = 4 instrs/thread
    if (pending) asm volatile("s_wait_asynccnt 0x4" ::: "memory");
    else         asm volatile("s_wait_asynccnt 0x0" ::: "memory");
}

// Issue this thread's 4 async B128 copies (one 64B W row-slice -> LDS).
// thread t: khalf = t>>7, row-slice rl = t&127 -> (jsel, gate, row);
// LDS bytes [t*64, t*64+64) == [khalf*8192 + rl*64 ...): dense staging.
__device__ __forceinline__ void stage_b(const char* __restrict__ wbase,
                                        unsigned rowStrideBytes, int j0a,
                                        unsigned kbBytes, unsigned ldsBase, int tid) {
    const int kh   = tid >> 7;
    const int rl   = tid & 127;
    const int jsel = rl >> 6;
    const int gate = (rl >> 4) & 3;
    const int row  = rl & 15;
    const char* g = wbase
        + (size_t)(gate * HDIM + j0a + jsel * 16 + row) * rowStrideBytes
        + kbBytes + kh * 64;
    const unsigned lds = ldsBase + (unsigned)tid * 64;
    #pragma unroll
    for (int u = 0; u < 4; ++u) {
        asm volatile("global_load_async_to_lds_b128 %0, %1, off"
                     :: "v"(lds + u * 16), "v"(g + u * 16) : "memory");
    }
}

// stage s (0..ST_N-1): h-part (64 K-elems of whh) or x-part (wih)
__device__ __forceinline__ void stage_s(int s, const __bf16* whh, const __bf16* wih,
                                        int j0a, unsigned ldsBase, int tid) {
    if (s < ST_H)
        stage_b((const char*)whh, HDIM * 2, j0a, (unsigned)(s * 128), ldsBase, tid);
    else
        stage_b((const char*)wih, EPAD * 2, j0a, (unsigned)((s - ST_H) * 128), ldsBase, tid);
}

// ---------------- prep kernels ---------------------------------------------

__global__ void k_convert_whh(const float* __restrict__ w, __bf16* __restrict__ o) {
    size_t i = (size_t)blockIdx.x * blockDim.x + threadIdx.x;
    if (i < (size_t)GDIM * HDIM) o[i] = (__bf16)w[i];
}

__global__ void k_convert_wih(const float* __restrict__ w, __bf16* __restrict__ o) {
    size_t i = (size_t)blockIdx.x * blockDim.x + threadIdx.x;
    if (i >= (size_t)GDIM * EPAD) return;
    int n = (int)(i / EPAD), e = (int)(i % EPAD);
    o[i] = (e < EDIM) ? (__bf16)w[(size_t)n * EDIM + e] : (__bf16)0.0f;
}

__global__ void k_bias(const float* __restrict__ bih, const float* __restrict__ bhh,
                       float* __restrict__ o) {
    int i = blockIdx.x * blockDim.x + threadIdx.x;
    if (i < GDIM) o[i] = bih[i] + bhh[i];
}

__global__ void k_embed(const int* __restrict__ tokens, const float* __restrict__ table,
                        __bf16* __restrict__ xb) {
    size_t i = (size_t)blockIdx.x * blockDim.x + threadIdx.x;
    if (i >= (size_t)T_LEN * BSZ * EPAD) return;
    int e  = (int)(i % EPAD);
    int tb = (int)(i / EPAD);            // t*B + b
    int tok = tokens[tb];
    xb[i] = (e < EDIM) ? (__bf16)table[(size_t)tok * EDIM + e] : (__bf16)0.0f;
}

__global__ void k_init_state(float* __restrict__ cf, float* __restrict__ cb,
                             __bf16* __restrict__ hf, __bf16* __restrict__ hb) {
    int i = blockIdx.x * blockDim.x + threadIdx.x;
    if (i >= BSZ * HDIM) return;
    cf[i] = 0.0f;
    cb[i] = 0.0f;
    hf[i] = (__bf16)0.0f;  hf[i + BSZ * HDIM] = (__bf16)0.0f;   // both parities
    hb[i] = (__bf16)0.0f;  hb[i + BSZ * HDIM] = (__bf16)0.0f;
}

// ---------------- fused LSTM step (both directions) -------------------------
// 128 blocks x 256 threads (8 waves).
// block  = (dir[1], jpair[6])  -> jtiles 2*jpair, 2*jpair+1
// wave w = (jsel = w>>2, mtile = w&3); each wave: 16x16 (b,j) block, 4 gates.

__global__ __launch_bounds__(256)
void k_lstm_step(const __bf16* __restrict__ whh,   // [8192][2048] bf16
                 const __bf16* __restrict__ wih,   // [8192][320]  bf16
                 const __bf16* __restrict__ xb,    // [128][64][320] bf16
                 const float*  __restrict__ bias,  // [8192]
                 float* __restrict__ c_fwd, float* __restrict__ c_bwd,   // [64][2048]
                 __bf16* __restrict__ hpp_fwd,     // [2][64][2048] ping-pong
                 __bf16* __restrict__ hpp_bwd,
                 const int* __restrict__ lens,
                 float* __restrict__ out,          // [64][4096]
                 int t)
{
    __shared__ char smem[3 * LDS_STAGE];

    const int tid   = threadIdx.x;
    const int lane  = tid & 31;
    const int w     = tid >> 5;           // wave in block: 0..7
    const int jsel  = w >> 2;             // 0/1
    const int mtile = w & 3;              // 0..3
    const int dir   = blockIdx.x >> 6;    // 0 = fwd, 1 = bwd
    const int jpair = blockIdx.x & 63;
    const int j0a   = jpair * 32;                 // first jtile origin (block)
    const int j0    = j0a + jsel * 16;            // this wave's jtile origin
    const int m0    = mtile * 16;

    const int ln15  = lane & 15;
    const int hi    = lane >> 4;
    const int abase = hi ? 8  : 0;        // A-fragment K offset (16-bit layout)

    __bf16* hpp = dir ? hpp_bwd : hpp_fwd;
    const __bf16* h_in  = hpp + (size_t)(t & 1) * (BSZ * HDIM);
    __bf16*       h_out = hpp + (size_t)((t + 1) & 1) * (BSZ * HDIM);
    float* cb = dir ? c_bwd : c_fwd;
    const int tx = dir ? (T_LEN - 1 - t) : t;

    // A row pointers (row-major bf16; two 16B loads per fragment)
    const __bf16* arow  = h_in + (size_t)(m0 + ln15) * HDIM;
    const __bf16* axrow = xb + (size_t)tx * BSZ * EPAD + (size_t)(m0 + ln15) * EPAD;

    // LDS base offset (low 32 bits of shared-aperture address = LDS offset)
    const unsigned ldsBuf0 = (unsigned)(size_t)(&smem[0]);

    // Per-wave B-fragment byte offset inside a stage buffer (khalf 0):
    // ((jsel*4 + gate)*16 + row)*64 + hi*16 ; gate stride 1024B, khalf stride 8192B
    const unsigned fragOff = (unsigned)(((jsel * 4) * 16 + ln15) * 64 + hi * 16);

    v8f acc0 = {}, acc1 = {}, acc2 = {}, acc3 = {};
    Frag aN0, aN1;   // prefetched A fragments for the *current* body

    // ---- prologue: 2 stages in flight + first A fragments ----------------
    stage_s(0, whh, wih, j0a, ldsBuf0 + 0 * LDS_STAGE, tid);
    stage_s(1, whh, wih, j0a, ldsBuf0 + 1 * LDS_STAGE, tid);
    aN0.q[0] = *(const uint4*)(arow + 0  + abase);
    aN0.q[1] = *(const uint4*)(arow + 16 + abase);
    aN1.q[0] = *(const uint4*)(arow + 32 + abase);
    aN1.q[1] = *(const uint4*)(arow + 48 + abase);

    // B-fragment loader (two aligned 16B LDS reads)
    auto ldB = [&](const char* p) {
        Frag b;
        b.q[0] = *(const uint4*)p;
        b.q[1] = *(const uint4*)(p + 32);
        return b;
    };

    // one pipeline body: consume stage s from curOff, issue stage s+2 to stgOff
    auto body = [&](int s, unsigned curOff, unsigned stgOff) {
        wait_async(s + 1 < ST_N);
        __syncthreads();                       // buffer curOff resident block-wide

        if (s + 2 < ST_N)
            stage_s(s + 2, whh, wih, j0a, ldsBuf0 + stgOff, tid);

        // rotate prefetched A in; prefetch A for body s+1
        Frag aC0 = aN0, aC1 = aN1;
        if (s + 1 < ST_N) {
            const int ns = s + 1;
            const __bf16* ap = (ns < ST_H) ? (arow + ns * 64)
                                           : (axrow + (ns - ST_H) * 64);
            aN0.q[0] = *(const uint4*)(ap + 0  + abase);
            aN0.q[1] = *(const uint4*)(ap + 16 + abase);
            aN1.q[0] = *(const uint4*)(ap + 32 + abase);
            aN1.q[1] = *(const uint4*)(ap + 48 + abase);
        }

        // 8 (khalf, gate) WMMAs, B loads pipelined ahead of use
        const char* p0 = smem + curOff + fragOff;          // khalf 0
        const char* p1 = p0 + 8192;                        // khalf 1
        Frag t0 = ldB(p0);
        Frag t1 = ldB(p0 + 1024);
        Frag t2 = ldB(p0 + 2048);
        acc0 = __builtin_amdgcn_wmma_f32_16x16x32_bf16(false, aC0.v, false, t0.v, (short)0, acc0, false, false);
        Frag t3 = ldB(p0 + 3072);
        acc1 = __builtin_amdgcn_wmma_f32_16x16x32_bf16(false, aC0.v, false, t1.v, (short)0, acc1, false, false);
        Frag t4 = ldB(p1);
        acc2 = __builtin_amdgcn_wmma_f32_16x16x32_bf16(false, aC0.v, false, t2.v, (short)0, acc2, false, false);
        Frag t5 = ldB(p1 + 1024);
        acc3 = __builtin_amdgcn_wmma_f32_16x16x32_bf16(false, aC0.v, false, t3.v, (short)0, acc3, false, false);
        Frag t6 = ldB(p1 + 2048);
        acc0 = __builtin_amdgcn_wmma_f32_16x16x32_bf16(false, aC1.v, false, t4.v, (short)0, acc0, false, false);
        Frag t7 = ldB(p1 + 3072);
        acc1 = __builtin_amdgcn_wmma_f32_16x16x32_bf16(false, aC1.v, false, t5.v, (short)0, acc1, false, false);
        acc2 = __builtin_amdgcn_wmma_f32_16x16x32_bf16(false, aC1.v, false, t6.v, (short)0, acc2, false, false);
        acc3 = __builtin_amdgcn_wmma_f32_16x16x32_bf16(false, aC1.v, false, t7.v, (short)0, acc3, false, false);

        // enforce DS-read / WMMA interleave (3 fragments prefetched ahead):
        // L L L W (L W)x5 W W  -- counts in individual ds_load instructions
        SGB(SGB_DSREAD, 6, 0);        // t0,t1,t2
        SGB(SGB_MATMUL, 1, 0);        // wmma(t0)
        SGB(SGB_DSREAD, 2, 0);        // t3
        SGB(SGB_MATMUL, 1, 0);        // wmma(t1)
        SGB(SGB_DSREAD, 2, 0);        // t4
        SGB(SGB_MATMUL, 1, 0);        // wmma(t2)
        SGB(SGB_DSREAD, 2, 0);        // t5
        SGB(SGB_MATMUL, 1, 0);        // wmma(t3)
        SGB(SGB_DSREAD, 2, 0);        // t6
        SGB(SGB_MATMUL, 1, 0);        // wmma(t4)
        SGB(SGB_DSREAD, 2, 0);        // t7
        SGB(SGB_MATMUL, 3, 0);        // wmma(t5..t7)
    };

    // 12 groups of 3 bodies (compile-time buffer offsets) + 1 tail = 37 stages
    #pragma unroll 1
    for (int g = 0; g < ST_N / 3; ++g) {
        const int s = g * 3;
        body(s + 0, 0 * LDS_STAGE, 2 * LDS_STAGE);
        body(s + 1, 1 * LDS_STAGE, 0 * LDS_STAGE);
        body(s + 2, 2 * LDS_STAGE, 1 * LDS_STAGE);
    }
    body(ST_N - 1, 0 * LDS_STAGE, 2 * LDS_STAGE);   // s = 36 (36 % 3 == 0)

    // ---- in-register LSTM cell update ------------------------------------
    // C/D layout: lane l, VGPR v -> M = v + (l>=16 ? 8 : 0), N = l&15.
    const int n = j0 + ln15;
    const float bi = bias[n];
    const float bf = bias[HDIM + n];
    const float bg = bias[2 * HDIM + n];
    const float bo = bias[3 * HDIM + n];

    #pragma unroll
    for (int v = 0; v < 8; ++v) {
        const int m = m0 + v + abase;
        float ig = sigmoidf_(acc0[v] + bi);
        float fg = sigmoidf_(acc1[v] + bf);
        float gg = tanhf(acc2[v] + bg);
        float og = sigmoidf_(acc3[v] + bo);
        float cprev = cb[(size_t)m * HDIM + n];
        float cnew  = fg * cprev + ig * gg;
        float h     = og * tanhf(cnew);
        cb[(size_t)m * HDIM + n] = cnew;
        h_out[(size_t)m * HDIM + n] = (__bf16)h;
        if (dir == 0) {
            if (lens[m] - 1 == t) out[(size_t)m * (2 * HDIM) + n] = h;
        } else if (t == T_LEN - 1) {
            out[(size_t)m * (2 * HDIM) + HDIM + n] = h;
        }
    }
}

// ---------------- host-side launch ------------------------------------------

extern "C" void kernel_launch(void* const* d_in, const int* in_sizes, int n_in,
                              void* d_out, int out_size, void* d_ws, size_t ws_size,
                              hipStream_t stream) {
    const int*   tokens = (const int*)  d_in[0];   // [128][64]
    const int*   lens   = (const int*)  d_in[1];   // [64]
    const float* table  = (const float*)d_in[2];   // [50000][300]
    const float* W_ih   = (const float*)d_in[3];   // [8192][300]
    const float* W_hh   = (const float*)d_in[4];   // [8192][2048]
    const float* b_ih   = (const float*)d_in[5];   // [8192]
    const float* b_hh   = (const float*)d_in[6];   // [8192]
    float* out = (float*)d_out;                    // [64][4096]

    char* ws = (char*)d_ws;
    size_t off = 0;
    auto alloc = [&](size_t bytes) { char* p = ws + off; off = (off + bytes + 255) & ~(size_t)255; return p; };

    __bf16* whh_bf = (__bf16*)alloc((size_t)GDIM * HDIM * 2);       // 33.5 MB
    __bf16* wih_bf = (__bf16*)alloc((size_t)GDIM * EPAD * 2);       //  5.2 MB
    __bf16* x_bf   = (__bf16*)alloc((size_t)T_LEN * BSZ * EPAD * 2);//  5.2 MB
    float*  bias   = (float*) alloc((size_t)GDIM * 4);
    float*  c_fwd  = (float*) alloc((size_t)BSZ * HDIM * 4);
    float*  c_bwd  = (float*) alloc((size_t)BSZ * HDIM * 4);
    __bf16* h_fwd  = (__bf16*)alloc((size_t)2 * BSZ * HDIM * 2);    // ping-pong
    __bf16* h_bwd  = (__bf16*)alloc((size_t)2 * BSZ * HDIM * 2);
    if (off > ws_size) return;  // workspace too small; bail deterministically

    // prep
    {
        size_t n = (size_t)GDIM * HDIM;
        k_convert_whh<<<(unsigned)((n + 255) / 256), 256, 0, stream>>>(W_hh, whh_bf);
    }
    {
        size_t n = (size_t)GDIM * EPAD;
        k_convert_wih<<<(unsigned)((n + 255) / 256), 256, 0, stream>>>(W_ih, wih_bf);
    }
    k_bias<<<(GDIM + 255) / 256, 256, 0, stream>>>(b_ih, b_hh, bias);
    {
        size_t n = (size_t)T_LEN * BSZ * EPAD;
        k_embed<<<(unsigned)((n + 255) / 256), 256, 0, stream>>>(tokens, table, x_bf);
    }
    k_init_state<<<(BSZ * HDIM + 255) / 256, 256, 0, stream>>>(c_fwd, c_bwd, h_fwd, h_bwd);

    // 128 sequential fused steps (fwd t, bwd T-1-t in one kernel)
    for (int t = 0; t < T_LEN; ++t) {
        k_lstm_step<<<128, 256, 0, stream>>>(whh_bf, wih_bf, x_bf, bias,
                                             c_fwd, c_bwd, h_fwd, h_bwd,
                                             lens, out, t);
    }
}
